// InICENODE_28355374088661
// MI455X (gfx1250) — compile-verified, matrix operand order checked by reading, not run on hardware
//
#include <hip/hip_runtime.h>

// ============================================================================
// ICENODE forward for MI455X (gfx1250): batch-tiled WMMA bf16 GEMMs.
// 64 workgroups x 256 threads (8 wave32). Each WG owns 16 batch rows (one
// WMMA M-tile) and walks the full T=32 sequential scan with all activations
// resident in LDS. Weights are pre-padded/converted to bf16 in d_ws.
// Activations use hardware V_TANH_F32 where available.
// ============================================================================

typedef __attribute__((ext_vector_type(16))) __bf16 v16bf;
typedef __attribute__((ext_vector_type(8)))  __bf16 v8bf;
typedef __attribute__((ext_vector_type(8)))  float  v8f;

union V16BF { v16bf v; v8bf h[2]; };
union V8F   { v8f v; float f[8]; };

#define NWAVES 8

// ---- problem constants ----
#define BB    1024
#define TT    32
#define OBS_N 40
#define OUTC  32

// ---- fast transcendentals: CDNA5 has V_TANH_F32 ----------------------------
__device__ __forceinline__ float fast_tanh(float x) {
#if __has_builtin(__builtin_amdgcn_tanhf)
  return __builtin_amdgcn_tanhf(x);
#elif __has_builtin(__builtin_amdgcn_tanh_f32)
  return __builtin_amdgcn_tanh_f32(x);
#else
  return tanhf(x);
#endif
}
__device__ __forceinline__ float fast_sigmoid(float x) {
#if __has_builtin(__builtin_amdgcn_tanhf) || __has_builtin(__builtin_amdgcn_tanh_f32)
  return 0.5f + 0.5f * fast_tanh(0.5f * x);
#else
  return 1.0f / (1.0f + __expf(-x));
#endif
}

// ---- weight pool (bf16) element offsets in d_ws; all padded:
//   dyn: [352x96][352x352][352x352][80x352]
//   obs: [128x32][48x128]   dx: [160x32][32x160]
//   gru: Wih [224x96]  Whh [224x96]
enum : int {
  W1_OFF  = 0,
  W2_OFF  = W1_OFF  + 352*96,
  W3_OFF  = W2_OFF  + 352*352,
  W4_OFF  = W3_OFF  + 352*352,
  OW1_OFF = W4_OFF  + 80*352,
  OW2_OFF = OW1_OFF + 128*32,
  XW1_OFF = OW2_OFF + 48*128,
  XW2_OFF = XW1_OFF + 160*32,
  WIH_OFF = XW2_OFF + 32*160,
  WHH_OFF = WIH_OFF + 224*96,
  WTOT    = WHH_OFF + 224*96,
};
// ---- bias pool (f32) element offsets (after bf16 pool) ----
enum : int {
  B1_OFF = 0, B2_OFF = 352, B3_OFF = 704, B4_OFF = 1056,
  OB1_OFF = 1136, OB2_OFF = 1264, XB1_OFF = 1312, XB2_OFF = 1472,
  BIH_OFF = 1504, BHH_OFF = 1728, BTOT = 1952,
};

// ---- LDS pool byte offsets (16B aligned). Scratch region is phase-unioned.
enum : int {
  OFF_S     = 0,       // f32 [16][80] state (cols>=70 stay 0)       5120
  OFF_H     = 5120,    // f32 [16]      h = dt/NSTEPS per row          64
  OFF_INTE  = 5184,    // bf16[16][16]  int_e padded                  512
  OFF_SCR   = 5696,
  // --- dyn phase ---
  OFF_DYNIN = OFF_SCR,            // bf16[16][96]                    3072
  OFF_ACTA  = OFF_SCR  + 3072,    // bf16[16][352]                  11264
  OFF_ACTB  = OFF_ACTA + 11264,   // bf16[16][352]                  11264
  OFF_KCUR  = OFF_ACTB + 11264,   // f32 [16][80]                    5120
  OFF_KACC  = OFF_KCUR + 5120,    // f32 [16][80]                    5120
  // --- gru / obs / dx phase (reuses scratch) ---
  OFF_GIN   = OFF_SCR,            // bf16 inputs (<=16*96)           3072
  OFF_HBF   = OFF_SCR + 3072,     // bf16[16][96] state as bf16      3072
  OFF_GI    = OFF_SCR + 6144,     // f32[16][224] / bf16 hidden     14336
  OFF_GH    = OFF_GI  + 14336,    // f32[16][224] / f32 pred        14336
  POOL      = OFF_KACC + 5120,    // 41536 bytes
};

// ----------------------------------------------------------------------------
// WMMA fragment load: 16 rows of bf16, row-major, stride S elements.
// Per ISA 7.12.2 (16-bit A 16x32): lane l -> row l&15; k-chunks
// [koff .. koff+7] and [16+koff .. 16+koff+7], koff = 8*(l>>4).
// B operand uses the mirrored layout with lane = output column n.
// ----------------------------------------------------------------------------
__device__ __forceinline__ v16bf load_frag(const __bf16* base, int stride,
                                           int kb, int lane) {
  const int r  = lane & 15;
  const int ko = (lane >> 4) << 3;
  const __bf16* p = base + r * stride + kb + ko;
  V16BF u;
  u.h[0] = *reinterpret_cast<const v8bf*>(p);
  u.h[1] = *reinterpret_cast<const v8bf*>(p + 16);
  return u.v;
}

// ----------------------------------------------------------------------------
// GEMM: OUT[16 x Np] = A(LDS bf16 [16 x KS]) @ W^T (bf16 [Np x KS]) + bias
// ACT: 0 = linear, store f32    1 = tanh, store bf16    2 = relu, store bf16
// Each wave owns N-tiles wave, wave+8, ... (16 cols per tile).
// ----------------------------------------------------------------------------
template <int ACT, int KS>
__device__ __forceinline__ void gemm_bf(const __bf16* A,
                                        const __bf16* __restrict__ W,
                                        const float*  __restrict__ bias,
                                        int Np, void* out, int ostride,
                                        int tid) {
  const int lane  = tid & 31;
  const int wave  = tid >> 5;
  const int ncol  = lane & 15;          // n within tile
  const int mbase = (lane >> 4) << 3;   // m = vgpr + mbase
  for (int nt = wave * 16; nt < Np; nt += 16 * NWAVES) {
    const float bval = bias[nt + ncol];
    V8F acc;
#pragma unroll
    for (int i = 0; i < 8; ++i) acc.f[i] = bval;
    const __bf16* Wrow = W + (size_t)nt * KS;
#pragma unroll
    for (int kb = 0; kb < KS; kb += 32) {
      v16bf a = load_frag(A, KS, kb, lane);
      v16bf b = load_frag(Wrow, KS, kb, lane);
      acc.v = __builtin_amdgcn_wmma_f32_16x16x32_bf16(
          false, a, false, b, (short)0, acc.v, false, false);
    }
    const int n = nt + ncol;
#pragma unroll
    for (int i = 0; i < 8; ++i) {
      const int m = mbase + i;
      const float x = acc.f[i];
      if (ACT == 0) {
        ((float*)out)[m * ostride + n] = x;
      } else if (ACT == 1) {
        ((__bf16*)out)[m * ostride + n] = (__bf16)fast_tanh(x);
      } else {
        ((__bf16*)out)[m * ostride + n] = (__bf16)fmaxf(x, 0.0f);
      }
    }
  }
}

// ---- dyn-MLP input: [ s + coef*h*kcur | int_e | 0 ] -> bf16 [16][96] -------
__device__ __forceinline__ void build_dynin(char* lds, int tid, float coef,
                                            bool usek) {
  const float*  s  = (const float*)(lds + OFF_S);
  const float*  h  = (const float*)(lds + OFF_H);
  const float*  kc = (const float*)(lds + OFF_KCUR);
  const __bf16* ie = (const __bf16*)(lds + OFF_INTE);
  __bf16* din = (__bf16*)(lds + OFF_DYNIN);
  for (int i = tid; i < 16 * 96; i += 256) {
    const int r = i / 96, c = i % 96;
    if (c < 70) {
      float v = s[r * 80 + c];
      if (usek) v += coef * h[r] * kc[r * 80 + c];
      din[i] = (__bf16)v;
    } else if (c < 85) {
      din[i] = ie[r * 16 + (c - 70)];
    } else {
      din[i] = (__bf16)0.0f;
    }
  }
}

// ---- one evaluation of the dynamics MLP: dynin -> kcur ---------------------
__device__ __forceinline__ void feval(char* lds, const __bf16* wp,
                                      const float* bp, int tid) {
  __bf16* din = (__bf16*)(lds + OFF_DYNIN);
  __bf16* a   = (__bf16*)(lds + OFF_ACTA);
  __bf16* b   = (__bf16*)(lds + OFF_ACTB);
  float*  kc  = (float*)(lds + OFF_KCUR);
  gemm_bf<1, 96 >(din, wp + W1_OFF, bp + B1_OFF, 352, a, 352, tid); __syncthreads();
  gemm_bf<1, 352>(a,   wp + W2_OFF, bp + B2_OFF, 352, b, 352, tid); __syncthreads();
  gemm_bf<1, 352>(b,   wp + W3_OFF, bp + B3_OFF, 352, a, 352, tid); __syncthreads();
  gemm_bf<0, 352>(a,   wp + W4_OFF, bp + B4_OFF, 80,  kc, 80, tid); __syncthreads();
}

__device__ __forceinline__ void kacc_step(char* lds, int tid, float w,
                                          bool first) {
  const float* kc = (const float*)(lds + OFF_KCUR);
  float* ka = (float*)(lds + OFF_KACC);
  for (int i = tid; i < 16 * 80; i += 256)
    ka[i] = (first ? 0.0f : ka[i]) + w * kc[i];
}

// ---- RK4, N_ODE_STEPS=2 ----------------------------------------------------
__device__ __forceinline__ void rk4_integrate(char* lds, const __bf16* wp,
                                              const float* bp, int tid) {
  float* s = (float*)(lds + OFF_S);
  const float* h = (const float*)(lds + OFF_H);
  const float* ka = (const float*)(lds + OFF_KACC);
  for (int step = 0; step < 2; ++step) {
    build_dynin(lds, tid, 0.0f, false); __syncthreads();
    feval(lds, wp, bp, tid);                       // k1
    kacc_step(lds, tid, 1.0f, true);
    build_dynin(lds, tid, 0.5f, true); __syncthreads();
    feval(lds, wp, bp, tid);                       // k2
    kacc_step(lds, tid, 2.0f, false);
    build_dynin(lds, tid, 0.5f, true); __syncthreads();
    feval(lds, wp, bp, tid);                       // k3
    kacc_step(lds, tid, 2.0f, false);
    build_dynin(lds, tid, 1.0f, true); __syncthreads();
    feval(lds, wp, bp, tid);                       // k4
    kacc_step(lds, tid, 1.0f, false); __syncthreads();
    for (int i = tid; i < 16 * 80; i += 256) {
      const int r = i / 80;
      s[i] += (h[r] * (1.0f / 6.0f)) * ka[i];
    }
    __syncthreads();
  }
}

// ============================================================================
// Main persistent kernel: gridDim.x = 64 (16 batch rows each), 256 threads.
// ============================================================================
__global__ __launch_bounds__(256) void icenode_main(
    const float* __restrict__ obs_times, const float* __restrict__ obs_vals,
    const float* __restrict__ obs_mask,  const float* __restrict__ dx_e0,
    const float* __restrict__ int_e,     const __bf16* __restrict__ wp,
    const float* __restrict__ bp,        float* __restrict__ out) {
  __shared__ __align__(16) char lds[POOL];
  const int tid  = threadIdx.x;
  const int row0 = blockIdx.x * 16;

  float*  s    = (float*)(lds + OFF_S);
  float*  hst  = (float*)(lds + OFF_H);
  __bf16* inte = (__bf16*)(lds + OFF_INTE);

  // init state: [0(15) | 0(25) | dx_e0(30) | 0 pad]
  for (int i = tid; i < 16 * 80; i += 256) {
    const int r = i / 80, c = i % 80;
    s[i] = (c >= 40 && c < 70) ? dx_e0[(size_t)(row0 + r) * 30 + (c - 40)]
                               : 0.0f;
  }
  for (int i = tid; i < 16 * 16; i += 256) {
    const int r = i / 16, c = i % 16;
    inte[i] = (__bf16)((c < 15) ? int_e[(size_t)(row0 + r) * 15 + c] : 0.0f);
  }
  float tprev = 0.0f;  // live only in lanes tid<16
  __syncthreads();

  for (int t = 0; t < TT; ++t) {
    if (tid < 16) {
      const float tt = obs_times[(size_t)(row0 + tid) * TT + t];
      hst[tid] = (tt - tprev) * 0.5f;  // dt / N_ODE_STEPS
      tprev = tt;
    }
    __syncthreads();

    rk4_integrate(lds, wp, bp, tid);

    // ---- obs MLP: relu([16x32] @ [128x32]^T) @ [48x128]^T ----
    __bf16* gin  = (__bf16*)(lds + OFF_GIN);
    __bf16* hbuf = (__bf16*)(lds + OFF_GI);   // bf16 hidden (phase union)
    float*  pred = (float*)(lds + OFF_GH);    // f32 preds  (phase union)
    for (int i = tid; i < 16 * 32; i += 256) {
      const int r = i / 32, c = i % 32;
      gin[i] = (__bf16)((c < 25) ? s[r * 80 + 15 + c] : 0.0f);
    }
    __syncthreads();
    gemm_bf<2, 32 >(gin,  wp + OW1_OFF, bp + OB1_OFF, 128, hbuf, 128, tid);
    __syncthreads();
    gemm_bf<0, 128>(hbuf, wp + OW2_OFF, bp + OB2_OFF, 48,  pred, 48, tid);
    __syncthreads();

    // ---- write preds; build GRU inputs ----
    for (int i = tid; i < 16 * OBS_N; i += 256) {
      const int r = i / OBS_N, c = i % OBS_N;
      out[((size_t)(row0 + r) * TT + t) * OBS_N + c] = pred[r * 48 + c];
    }
    __bf16* hbf = (__bf16*)(lds + OFF_HBF);
    for (int i = tid; i < 16 * 96; i += 256) {
      const int r = i / 96, c = i % 96;
      float v = 0.0f;
      const size_t bt = ((size_t)(row0 + r) * TT + t) * OBS_N;
      if (c < 40) {
        const float m = obs_mask[bt + c];
        v = (m > 0.0f) ? (pred[r * 48 + c] - obs_vals[bt + c]) : 0.0f;
      } else if (c < 80) {
        v = obs_mask[bt + (c - 40)];
      }
      gin[i] = (__bf16)v;
      hbf[i] = (__bf16)((c < 70) ? s[r * 80 + c] : 0.0f);
    }
    __syncthreads();

    // ---- GRU: gi = gru_in @ Wih^T + bih ; gh = h @ Whh^T + bhh ----
    float* gi = (float*)(lds + OFF_GI);
    float* gh = (float*)(lds + OFF_GH);
    gemm_bf<0, 96>(gin, wp + WIH_OFF, bp + BIH_OFF, 224, gi, 224, tid);
    gemm_bf<0, 96>(hbf, wp + WHH_OFF, bp + BHH_OFF, 224, gh, 224, tid);
    __syncthreads();
    for (int i = tid; i < 16 * 70; i += 256) {
      const int r = i / 70, c = i % 70;
      const float ir = gi[r * 224 + c],       hr = gh[r * 224 + c];
      const float iz = gi[r * 224 + 70 + c],  hz = gh[r * 224 + 70 + c];
      const float in = gi[r * 224 + 140 + c], hn = gh[r * 224 + 140 + c];
      const float rg = fast_sigmoid(ir + hr);
      const float zg = fast_sigmoid(iz + hz);
      const float ng = fast_tanh(in + rg * hn);
      s[r * 80 + c] = (1.0f - zg) * ng + zg * s[r * 80 + c];
    }
    __syncthreads();
  }

  // ---- final integrate to T_HOURS ----
  if (tid < 16) hst[tid] = (72.0f - tprev) * 0.5f;
  __syncthreads();
  rk4_integrate(lds, wp, bp, tid);

  // ---- dx MLP: relu([16x32] @ [160x32]^T) @ [32x160]^T ----
  __bf16* gin  = (__bf16*)(lds + OFF_GIN);
  __bf16* hbuf = (__bf16*)(lds + OFF_GI);
  float*  pred = (float*)(lds + OFF_GH);
  for (int i = tid; i < 16 * 32; i += 256) {
    const int r = i / 32, c = i % 32;
    gin[i] = (__bf16)((c < 30) ? s[r * 80 + 40 + c] : 0.0f);
  }
  __syncthreads();
  gemm_bf<2, 32 >(gin,  wp + XW1_OFF, bp + XB1_OFF, 160, hbuf, 160, tid);
  __syncthreads();
  gemm_bf<0, 160>(hbuf, wp + XW2_OFF, bp + XB2_OFF, 32,  pred, 32, tid);
  __syncthreads();
  for (int i = tid; i < 16 * OUTC; i += 256) {
    const int r = i / OUTC, c = i % OUTC;
    out[(size_t)BB * TT * OBS_N + (size_t)(row0 + r) * OUTC + c] =
        pred[r * OUTC + c];
  }
}

// ============================================================================
// Prep: pad + convert fp32 weights -> bf16 [doutp x dinp], biases -> f32 pad.
// ============================================================================
__global__ void pad_w(const float* __restrict__ src, __bf16* __restrict__ dst,
                      int dout, int din, int dinp, int total) {
  const int i = blockIdx.x * 256 + threadIdx.x;
  if (i >= total) return;
  const int n = i / dinp, k = i % dinp;
  dst[i] = (__bf16)((n < dout && k < din) ? src[(size_t)n * din + k] : 0.0f);
}
__global__ void pad_b(const float* __restrict__ src, float* __restrict__ dst,
                      int d, int total) {
  const int i = blockIdx.x * 256 + threadIdx.x;
  if (i >= total) return;
  dst[i] = (i < d) ? src[i] : 0.0f;
}

extern "C" void kernel_launch(void* const* d_in, const int* in_sizes, int n_in,
                              void* d_out, int out_size, void* d_ws,
                              size_t ws_size, hipStream_t stream) {
  (void)in_sizes; (void)n_in; (void)out_size; (void)ws_size;
  // setup_inputs order (tuples flattened in order):
  // 0 obs_times 1 obs_vals 2 obs_mask 3 dx_e0 4 int_e
  // 5-8 dyn_Ws  9-12 dyn_bs  13-14 obs_Ws 15-16 obs_bs
  // 17-18 dx_Ws 19-20 dx_bs  21 gru_Wih 22 gru_Whh 23 gru_bih 24 gru_bhh
  __bf16* wp = (__bf16*)d_ws;
  float*  bp = (float*)((char*)d_ws + (size_t)WTOT * 2);

  auto padw = [&](const void* src, int off, int dout, int din, int dinp,
                  int doutp) {
    const int total = doutp * dinp;
    pad_w<<<(total + 255) / 256, 256, 0, stream>>>((const float*)src,
                                                   wp + off, dout, din, dinp,
                                                   total);
  };
  auto padb = [&](const void* src, int off, int d, int dp) {
    pad_b<<<(dp + 255) / 256, 256, 0, stream>>>((const float*)src, bp + off, d,
                                                dp);
  };
  padw(d_in[5],  W1_OFF,  350, 85,  96,  352);
  padw(d_in[6],  W2_OFF,  350, 350, 352, 352);
  padw(d_in[7],  W3_OFF,  350, 350, 352, 352);
  padw(d_in[8],  W4_OFF,  70,  350, 352, 80);
  padw(d_in[13], OW1_OFF, 125, 25,  32,  128);
  padw(d_in[14], OW2_OFF, 40,  125, 128, 48);
  padw(d_in[17], XW1_OFF, 150, 30,  32,  160);
  padw(d_in[18], XW2_OFF, 32,  150, 160, 32);
  padw(d_in[21], WIH_OFF, 210, 80,  96,  224);
  padw(d_in[22], WHH_OFF, 210, 70,  96,  224);

  padb(d_in[9],  B1_OFF,  350, 352);
  padb(d_in[10], B2_OFF,  350, 352);
  padb(d_in[11], B3_OFF,  350, 352);
  padb(d_in[12], B4_OFF,  70,  80);
  padb(d_in[15], OB1_OFF, 125, 128);
  padb(d_in[16], OB2_OFF, 40,  48);
  padb(d_in[19], XB1_OFF, 150, 160);
  padb(d_in[20], XB2_OFF, 32,  32);
  padb(d_in[23], BIH_OFF, 210, 224);
  padb(d_in[24], BHH_OFF, 210, 224);

  icenode_main<<<dim3(BB / 16), dim3(256), 0, stream>>>(
      (const float*)d_in[0], (const float*)d_in[1], (const float*)d_in[2],
      (const float*)d_in[3], (const float*)d_in[4], wp, bp, (float*)d_out);
}
